// CyberMoE_32315333935485
// MI455X (gfx1250) — matrix operand order repaired and checked by reading, not compiled, not run on hardware
//
#include <hip/hip_runtime.h>
#include <hip/hip_bf16.h>
#include <math.h>

// Problem constants
#define Bb   256
#define Ss   128
#define Hh   768
#define NHh  8
#define HDd  96
#define QKVN (3*Hh)      // 2304
#define Mm   (Bb*Ss)     // 32768

typedef __attribute__((ext_vector_type(16))) __bf16 v16bf;
typedef __attribute__((ext_vector_type(8)))  float  v8f;

// Raw LDS byte offset of a __shared__ object (addrspace(3) ptrtoint).
typedef __attribute__((address_space(3))) const unsigned char lds_cbyte;
__device__ __forceinline__ unsigned lds_addr_of(const void* p) {
    return (unsigned)(unsigned long long)(lds_cbyte*)p;
}

// Async global -> LDS 16B copy (CDNA5, tracked with ASYNCcnt)
__device__ __forceinline__ void async_copy_b128(unsigned lds_off, const void* gptr) {
    asm volatile("global_load_async_to_lds_b128 %0, %1, off"
                 :: "v"(lds_off), "v"((unsigned long long)gptr)
                 : "memory");
}
__device__ __forceinline__ void async_wait0() {
    asm volatile("s_wait_asynccnt 0x0" ::: "memory");
}

// ---------------------------------------------------------------------------
// fp32 -> bf16 conversion (grid-stride)
// ---------------------------------------------------------------------------
__global__ void cvt_f32_bf16(const float* __restrict__ x, __bf16* __restrict__ y, size_t n) {
    size_t i = (size_t)blockIdx.x * blockDim.x + threadIdx.x;
    size_t stride = (size_t)gridDim.x * blockDim.x;
    for (; i < n; i += stride) y[i] = (__bf16)x[i];
}

// ---------------------------------------------------------------------------
// QKV GEMM: C[M,N] = A[M,K] @ W[N,K]^T + bias, bf16 in, bf16 out, f32 accum
// WG tile 128x64, 8 waves as 4(M) x 2(N), each wave 32x32 (2x2 WMMA tiles)
// Tiles staged with global_load_async_to_lds_b128.
// ---------------------------------------------------------------------------
#define TM 128
#define TN 64
#define TK 32
#define ASTR 40
#define BSTR 40

__global__ __launch_bounds__(256) void gemm_qkv_bf16(
    const __bf16* __restrict__ A,    // [M,K]
    const __bf16* __restrict__ W,    // [N,K]
    const float*  __restrict__ bias, // [N]
    __bf16* __restrict__ C,          // [M,N]
    int M, int N, int K)
{
    __shared__ __align__(16) __bf16 sA[TM * ASTR]; // 10240 B
    __shared__ __align__(16) __bf16 sB[TN * BSTR]; //  5120 B

    const int tid  = threadIdx.x;
    const int wave = tid >> 5;
    const int lane = tid & 31;
    const int l16  = lane & 15;
    const int hi   = lane >> 4;
    const int m0   = blockIdx.y * TM;
    const int n0   = blockIdx.x * TN;
    const int wm   = (wave >> 1) * 32;
    const int wn   = (wave & 1) * 32;

    const unsigned sA_base = lds_addr_of(sA);
    const unsigned sB_base = lds_addr_of(sB);

    // Per-thread staging coordinates (fixed for all K-steps)
    const int arow0 = tid >> 2,          akc0 = (tid & 3) << 3;          // chunk tid
    const int arow1 = (tid + 256) >> 2,  akc1 = (tid & 3) << 3;          // chunk tid+256
    const int brow  = tid >> 2,          bkc  = (tid & 3) << 3;
    const unsigned la0 = sA_base + (unsigned)(arow0 * ASTR + akc0) * 2;
    const unsigned la1 = sA_base + (unsigned)(arow1 * ASTR + akc1) * 2;
    const unsigned lb  = sB_base + (unsigned)(brow  * BSTR + bkc ) * 2;

    v8f acc[2][2];
    const v8f vzero = {0.f,0.f,0.f,0.f,0.f,0.f,0.f,0.f};
    for (int i = 0; i < 2; ++i)
        for (int j = 0; j < 2; ++j) acc[i][j] = vzero;

    for (int kt = 0; kt < K; kt += TK) {
        // Async stage A tile (2 x 16B per thread) and B tile (1 x 16B per thread)
        async_copy_b128(la0, A + (size_t)(m0 + arow0) * K + kt + akc0);
        async_copy_b128(la1, A + (size_t)(m0 + arow1) * K + kt + akc1);
        async_copy_b128(lb,  W + (size_t)(n0 + brow ) * K + kt + bkc );
        async_wait0();
        __syncthreads();

        if (kt + TK < K) {
            __builtin_prefetch(A + (size_t)(m0 + (tid >> 1)) * K + kt + TK, 0, 1);
            if (tid < TN) __builtin_prefetch(W + (size_t)(n0 + tid) * K + kt + TK, 0, 1);
        }

        #pragma unroll
        for (int i = 0; i < 2; ++i) {
            union { v16bf v; uint4 u[2]; } fa;
            const __bf16* ap = &sA[(wm + 16 * i + l16) * ASTR + hi * 8];
            fa.u[0] = *(const uint4*)(ap);
            fa.u[1] = *(const uint4*)(ap + 16);
            #pragma unroll
            for (int j = 0; j < 2; ++j) {
                union { v16bf v; uint4 u[2]; } fb;
                const __bf16* bp = &sB[(wn + 16 * j + l16) * BSTR + hi * 8];
                fb.u[0] = *(const uint4*)(bp);
                fb.u[1] = *(const uint4*)(bp + 16);
                acc[i][j] = __builtin_amdgcn_wmma_f32_16x16x32_bf16(
                    false, fa.v, false, fb.v, (short)0, acc[i][j], false, false);
            }
        }
        __syncthreads();
    }

    // Epilogue: C layout -> vgpr r: m = r + 8*hi ; n = l16
    for (int i = 0; i < 2; ++i)
        for (int j = 0; j < 2; ++j) {
            int n = n0 + wn + 16 * j + l16;
            float bn = bias[n];
            #pragma unroll
            for (int r = 0; r < 8; ++r) {
                int m = m0 + wm + 16 * i + r + 8 * hi;
                C[(size_t)m * N + n] = (__bf16)(acc[i][j][r] + bn);
            }
        }
}

// ---------------------------------------------------------------------------
// Attention per (b,h): scores = QK^T/sqrt(HD), softmax, P@V.
// Directly reduces sum over s into ao_mean[b, h*HD .. ] (un-normalized sum).
// LDS (61440 B): [0,34816): K tile (stride 104) overlaid later by P (stride 136)
//                [34816,60928): V^T (96 x 136)
//                [60928,61312): 96-float column accumulator
// K tile staged with global_load_async_to_lds_b128.
// ---------------------------------------------------------------------------
#define KSTR 104
#define PSTR 136
#define VTSTR 136

__global__ __launch_bounds__(256) void attn_kernel(
    const __bf16* __restrict__ qkv,  // [B*S, 2304]
    float* __restrict__ ao_mean)     // [B, 768] (sum over s)
{
    __shared__ __align__(16) unsigned char smem[34816 + 26112 + 512];
    __bf16* sK   = (__bf16*)smem;
    __bf16* sP   = (__bf16*)smem;
    __bf16* sVt  = (__bf16*)(smem + 34816);
    float*  sAcc = (float*)(smem + 34816 + 26112);

    const int tid  = threadIdx.x;
    const int wave = tid >> 5;
    const int lane = tid & 31;
    const int l16  = lane & 15;
    const int hi   = lane >> 4;
    const int bb   = blockIdx.x >> 3;
    const int h    = blockIdx.x & 7;

    const unsigned sK_base = lds_addr_of(smem);

    if (tid < HDd) sAcc[tid] = 0.f;

    // Stage K asynchronously: 128 rows x 96 bf16 = 1536 x 16B chunks, 6/thread
    #pragma unroll
    for (int i = 0; i < 6; ++i) {
        int c  = tid + 256 * i;
        int s  = c / 12;
        int kc = (c % 12) * 8;
        async_copy_b128(sK_base + (unsigned)(s * KSTR + kc) * 2,
                        qkv + (size_t)(bb * Ss + s) * QKVN + Hh + h * HDd + kc);
    }
    // Stage V transposed (element scatter, classic path): sVt[d][s]
    for (int c = tid; c < 3072; c += 256) {
        int s  = c / 24;
        int dc = (c % 24) * 4;
        union { uint2 u; __bf16 hv[4]; } t;
        t.u = *(const uint2*)(qkv + (size_t)(bb * Ss + s) * QKVN + 2 * Hh + h * HDd + dc);
        sVt[(dc + 0) * VTSTR + s] = t.hv[0];
        sVt[(dc + 1) * VTSTR + s] = t.hv[1];
        sVt[(dc + 2) * VTSTR + s] = t.hv[2];
        sVt[(dc + 3) * VTSTR + s] = t.hv[3];
    }
    async_wait0();
    __syncthreads();

    // Phase 1: scores tile (16 rows per wave) x 128 cols
    v8f accS[8];
    const v8f vzero = {0.f,0.f,0.f,0.f,0.f,0.f,0.f,0.f};
    for (int j = 0; j < 8; ++j) accS[j] = vzero;

    const int qrow = wave * 16 + l16;
    const __bf16* qbase = qkv + (size_t)(bb * Ss + qrow) * QKVN + h * HDd;

    for (int kt = 0; kt < HDd; kt += 32) {
        union { v16bf v; uint4 u[2]; } fa;
        fa.u[0] = *(const uint4*)(qbase + kt + hi * 8);
        fa.u[1] = *(const uint4*)(qbase + kt + hi * 8 + 16);
        #pragma unroll
        for (int j = 0; j < 8; ++j) {
            union { v16bf v; uint4 u[2]; } fb;
            const __bf16* bp = &sK[(16 * j + l16) * KSTR + kt + hi * 8];
            fb.u[0] = *(const uint4*)(bp);
            fb.u[1] = *(const uint4*)(bp + 16);
            accS[j] = __builtin_amdgcn_wmma_f32_16x16x32_bf16(
                false, fa.v, false, fb.v, (short)0, accS[j], false, false);
        }
    }

    // Softmax over 128 cols per row (row spread over 16 lanes x 8 accums)
    const float scale = 0.102062072615966f; // 1/sqrt(96)
    #pragma unroll
    for (int r = 0; r < 8; ++r) {
        float mx = -1e30f;
        #pragma unroll
        for (int j = 0; j < 8; ++j) mx = fmaxf(mx, accS[j][r] * scale);
        for (int m = 1; m < 16; m <<= 1) mx = fmaxf(mx, __shfl_xor(mx, m, 16));
        float sum = 0.f;
        #pragma unroll
        for (int j = 0; j < 8; ++j) {
            float e = __expf(accS[j][r] * scale - mx);
            accS[j][r] = e;
            sum += e;
        }
        for (int m = 1; m < 16; m <<= 1) sum += __shfl_xor(sum, m, 16);
        float inv = 1.f / sum;
        #pragma unroll
        for (int j = 0; j < 8; ++j) accS[j][r] *= inv;
    }
    __syncthreads();  // all waves done reading sK

    // Write probs (bf16) over the K-tile region with PSTR stride
    #pragma unroll
    for (int j = 0; j < 8; ++j)
        #pragma unroll
        for (int r = 0; r < 8; ++r)
            sP[(wave * 16 + r + 8 * hi) * PSTR + 16 * j + l16] = (__bf16)accS[j][r];
    __syncthreads();

    // Phase 2: out = P(16x128) @ V(128x96)
    v8f accO[6];
    for (int j = 0; j < 6; ++j) accO[j] = vzero;
    for (int kt = 0; kt < Ss; kt += 32) {
        union { v16bf v; uint4 u[2]; } fa;
        const __bf16* ap = &sP[(wave * 16 + l16) * PSTR + kt + hi * 8];
        fa.u[0] = *(const uint4*)(ap);
        fa.u[1] = *(const uint4*)(ap + 16);
        #pragma unroll
        for (int j = 0; j < 6; ++j) {
            union { v16bf v; uint4 u[2]; } fb;
            const __bf16* bp = &sVt[(16 * j + l16) * VTSTR + kt + hi * 8];
            fb.u[0] = *(const uint4*)(bp);
            fb.u[1] = *(const uint4*)(bp + 16);
            accO[j] = __builtin_amdgcn_wmma_f32_16x16x32_bf16(
                false, fa.v, false, fb.v, (short)0, accO[j], false, false);
        }
    }

    // Column sums over this wave's 16 rows -> LDS atomic accumulate
    #pragma unroll
    for (int j = 0; j < 6; ++j) {
        float cs = 0.f;
        #pragma unroll
        for (int r = 0; r < 8; ++r) cs += accO[j][r];
        atomicAdd(&sAcc[16 * j + l16], cs);
    }
    __syncthreads();

    if (tid < HDd) ao_mean[(size_t)bb * Hh + h * HDd + tid] = sAcc[tid];
}

// ---------------------------------------------------------------------------
// Row-parallel dense: Y[r,n] = alpha*(X[r,:K]·W[n,:K]) + bias[n],
// optional LayerNorm(g,be) + exact GELU. One block per row.
// dyn smem: K + N + 256 floats
// ---------------------------------------------------------------------------
__global__ __launch_bounds__(256) void row_linear(
    const float* __restrict__ X, int ldx, float alpha,
    const float* __restrict__ W, const float* __restrict__ bias,
    const float* __restrict__ g, const float* __restrict__ be,
    float* __restrict__ Y, int K, int N, int fuse_ln_gelu)
{
    extern __shared__ float smf[];
    float* sx  = smf;
    float* sy  = smf + K;
    float* red = smf + K + N;
    const int r = blockIdx.x;
    const int tid = threadIdx.x;

    for (int i = tid; i < K; i += 256) sx[i] = X[(size_t)r * ldx + i];
    __syncthreads();

    const float4* sx4 = (const float4*)sx;
    for (int n = tid; n < N; n += 256) {
        const float4* wr = (const float4*)(W + (size_t)n * K);
        float s = 0.f;
        for (int k4 = 0; k4 < (K >> 2); ++k4) {
            float4 a = sx4[k4], w4 = wr[k4];
            s += a.x * w4.x + a.y * w4.y + a.z * w4.z + a.w * w4.w;
        }
        sy[n] = alpha * s + bias[n];
    }
    __syncthreads();

    if (!fuse_ln_gelu) {
        for (int n = tid; n < N; n += 256) Y[(size_t)r * N + n] = sy[n];
        return;
    }

    float ps = 0.f;
    for (int n = tid; n < N; n += 256) ps += sy[n];
    red[tid] = ps; __syncthreads();
    for (int s = 128; s > 0; s >>= 1) { if (tid < s) red[tid] += red[tid + s]; __syncthreads(); }
    float mean = red[0] / N; __syncthreads();

    float pv = 0.f;
    for (int n = tid; n < N; n += 256) { float d = sy[n] - mean; pv += d * d; }
    red[tid] = pv; __syncthreads();
    for (int s = 128; s > 0; s >>= 1) { if (tid < s) red[tid] += red[tid + s]; __syncthreads(); }
    float rstd = rsqrtf(red[0] / N + 1e-5f);

    for (int n = tid; n < N; n += 256) {
        float y = (sy[n] - mean) * rstd * g[n] + be[n];
        Y[(size_t)r * N + n] = 0.5f * y * (1.f + erff(y * 0.70710678118654752f));
    }
}

// ---------------------------------------------------------------------------
// Final MoE head: gating softmax, top-2, expert logits, domain logits.
// One block (32 threads) per batch element.
// ---------------------------------------------------------------------------
__global__ __launch_bounds__(32) void finalize_kernel(
    const float* __restrict__ rv,  const float* __restrict__ w_r2, const float* __restrict__ b_r2,
    const float* __restrict__ dv,  const float* __restrict__ w_d2, const float* __restrict__ b_d2,
    const float* __restrict__ hs,  const float* __restrict__ w_e,  const float* __restrict__ b_e,
    float* __restrict__ out)
{
    const int b = blockIdx.x;
    const int t = threadIdx.x;
    __shared__ float gl[5], dl[5], ae[10], probs[5];

    if (t < 5) {
        float s = 0.f;
        for (int k = 0; k < 384; ++k) s += rv[(size_t)b * 384 + k] * w_r2[t * 384 + k];
        gl[t] = s + b_r2[t];
    } else if (t < 10) {
        int j = t - 5;
        float s = 0.f;
        for (int k = 0; k < 384; ++k) s += dv[(size_t)b * 384 + k] * w_d2[j * 384 + k];
        dl[j] = s + b_d2[j];
    } else if (t < 20) {
        int j = t - 10;  // j = e*2 + l
        float s = 0.f;
        const float* cls = hs + (size_t)b * Ss * Hh;
        for (int k = 0; k < Hh; ++k) s += cls[k] * w_e[(size_t)j * Hh + k];
        ae[j] = s + b_e[j];
    }
    __syncthreads();

    if (t == 0) {
        float mx = gl[0];
        for (int e = 1; e < 5; ++e) mx = fmaxf(mx, gl[e]);
        float sum = 0.f;
        for (int e = 0; e < 5; ++e) { probs[e] = __expf(gl[e] - mx); sum += probs[e]; }
        float inv = 1.f / sum;
        for (int e = 0; e < 5; ++e) probs[e] *= inv;

        // outputs: final[512] | gating[1280] | expert[2560] | domain[1280]
        for (int e = 0; e < 5; ++e) out[512 + b * 5 + e]  = probs[e];
        for (int e = 0; e < 5; ++e) out[4352 + b * 5 + e] = dl[e];

        int i1 = 0;
        for (int e = 1; e < 5; ++e) if (probs[e] > probs[i1]) i1 = e;
        int i2 = -1;
        for (int e = 0; e < 5; ++e) if (e != i1 && (i2 < 0 || probs[e] > probs[i2])) i2 = e;
        float p1 = probs[i1], p2 = probs[i2];
        float pinv = 1.f / (p1 + p2);
        p1 *= pinv; p2 *= pinv;

        for (int l = 0; l < 2; ++l)
            out[b * 2 + l] = p1 * ae[i1 * 2 + l] + p2 * ae[i2 * 2 + l];

        for (int e = 0; e < 5; ++e)
            for (int l = 0; l < 2; ++l)
                out[1792 + b * 10 + e * 2 + l] = (e == i1 || e == i2) ? ae[e * 2 + l] : 0.f;
    }
}

// ---------------------------------------------------------------------------
// Host launch
// ---------------------------------------------------------------------------
extern "C" void kernel_launch(void* const* d_in, const int* in_sizes, int n_in,
                              void* d_out, int out_size, void* d_ws, size_t ws_size,
                              hipStream_t stream) {
    (void)in_sizes; (void)n_in; (void)out_size; (void)ws_size;

    const float* hs    = (const float*)d_in[0];
    const float* w_qkv = (const float*)d_in[1];
    const float* b_qkv = (const float*)d_in[2];
    const float* w_out = (const float*)d_in[3];
    const float* b_out = (const float*)d_in[4];
    const float* w_f1  = (const float*)d_in[5];
    const float* b_f1  = (const float*)d_in[6];
    const float* g_f1  = (const float*)d_in[7];
    const float* be_f1 = (const float*)d_in[8];
    const float* w_f2  = (const float*)d_in[9];
    const float* b_f2  = (const float*)d_in[10];
    const float* g_f2  = (const float*)d_in[11];
    const float* be_f2 = (const float*)d_in[12];
    const float* w_c   = (const float*)d_in[13];
    const float* b_c   = (const float*)d_in[14];
    const float* g_c   = (const float*)d_in[15];
    const float* be_c  = (const float*)d_in[16];
    const float* w_r1  = (const float*)d_in[17];
    const float* b_r1  = (const float*)d_in[18];
    const float* g_r1  = (const float*)d_in[19];
    const float* be_r1 = (const float*)d_in[20];
    const float* w_r2  = (const float*)d_in[21];
    const float* b_r2  = (const float*)d_in[22];
    const float* w_d1  = (const float*)d_in[23];
    const float* b_d1  = (const float*)d_in[24];
    const float* g_d1  = (const float*)d_in[25];
    const float* be_d1 = (const float*)d_in[26];
    const float* w_d2  = (const float*)d_in[27];
    const float* b_d2  = (const float*)d_in[28];
    const float* w_e   = (const float*)d_in[29];
    const float* b_e   = (const float*)d_in[30];
    float* out = (float*)d_out;

    // Workspace carve-up (256B aligned)
    char* ws = (char*)d_ws;
    size_t off = 0;
    auto carve = [&](size_t bytes) { size_t o = off; off = (off + bytes + 255) & ~(size_t)255; return o; };
    __bf16* hsb     = (__bf16*)(ws + carve((size_t)Mm * Hh * 2));        // 50 MB
    __bf16* wqkvb   = (__bf16*)(ws + carve((size_t)QKVN * Hh * 2));      // 3.5 MB
    __bf16* qkvb    = (__bf16*)(ws + carve((size_t)Mm * QKVN * 2));      // 151 MB
    float*  ao_mean = (float*)(ws + carve((size_t)Bb * Hh * 4));
    float*  seq     = (float*)(ws + carve((size_t)Bb * Hh * 4));
    float*  f1b     = (float*)(ws + carve((size_t)Bb * 2 * Hh * 4));
    float*  f2b     = (float*)(ws + carve((size_t)Bb * 2 * Hh * 4));
    float*  ctx     = (float*)(ws + carve((size_t)Bb * Hh * 4));
    float*  rv      = (float*)(ws + carve((size_t)Bb * 384 * 4));
    float*  dv      = (float*)(ws + carve((size_t)Bb * 384 * 4));

    // 1) bf16 conversions
    cvt_f32_bf16<<<4096, 256, 0, stream>>>(hs, hsb, (size_t)Mm * Hh);
    cvt_f32_bf16<<<1024, 256, 0, stream>>>(w_qkv, wqkvb, (size_t)QKVN * Hh);

    // 2) QKV GEMM (dominant: 116 GFLOP, bf16 WMMA, async LDS staging)
    dim3 ggrid(QKVN / TN, Mm / TM);
    gemm_qkv_bf16<<<ggrid, 256, 0, stream>>>(hsb, wqkvb, b_qkv, qkvb, Mm, QKVN, Hh);

    // 3) Attention per (b,h), fused mean-over-s reduction
    attn_kernel<<<Bb * NHh, 256, 0, stream>>>(qkvb, ao_mean);

    // 4) seq_repr = (sum_s attn_out)/S @ w_out^T + b_out  (mean/linear commuted)
    row_linear<<<Bb, 256, (Hh + Hh + 256) * 4, stream>>>(
        ao_mean, Hh, 1.0f / Ss, w_out, b_out, nullptr, nullptr, seq, Hh, Hh, 0);

    // 5) MLP chain with fused LN + exact GELU
    row_linear<<<Bb, 256, (Hh + 2 * Hh + 256) * 4, stream>>>(
        seq, Hh, 1.f, w_f1, b_f1, g_f1, be_f1, f1b, Hh, 2 * Hh, 1);
    row_linear<<<Bb, 256, (2 * Hh + 2 * Hh + 256) * 4, stream>>>(
        f1b, 2 * Hh, 1.f, w_f2, b_f2, g_f2, be_f2, f2b, 2 * Hh, 2 * Hh, 1);
    row_linear<<<Bb, 256, (2 * Hh + Hh + 256) * 4, stream>>>(
        f2b, 2 * Hh, 1.f, w_c, b_c, g_c, be_c, ctx, 2 * Hh, Hh, 1);
    row_linear<<<Bb, 256, (Hh + 384 + 256) * 4, stream>>>(
        ctx, Hh, 1.f, w_r1, b_r1, g_r1, be_r1, rv, Hh, 384, 1);
    // domain branch from cls (row stride S*H picks hidden_state[:,0,:])
    row_linear<<<Bb, 256, (Hh + 384 + 256) * 4, stream>>>(
        hs, Ss * Hh, 1.f, w_d1, b_d1, g_d1, be_d1, dv, Hh, 384, 1);

    // 6) gating softmax + top-2 + expert/final/domain outputs
    finalize_kernel<<<Bb, 32, 0, stream>>>(rv, w_r2, b_r2, dv, w_d2, b_d2,
                                           hs, w_e, b_e, out);
}
